// StockPredictionModel_32212254720769
// MI455X (gfx1250) — matrix operand order repaired and checked by reading, not compiled, not run on hardware
//
#include <hip/hip_runtime.h>
#include <hip/hip_bf16.h>

typedef __attribute__((ext_vector_type(16))) _Float16 v16h;
typedef __attribute__((ext_vector_type(8)))  _Float16 v8h;
typedef __attribute__((ext_vector_type(8)))  float    v8f;
typedef __attribute__((ext_vector_type(4)))  unsigned u32x4;
typedef __attribute__((ext_vector_type(8)))  int      i32x8;
typedef __attribute__((ext_vector_type(4)))  int      i32x4;

#define TPB 256

__device__ __forceinline__ v16h cat8(v8h lo, v8h hi) {
    v16h r;
#pragma unroll
    for (int i = 0; i < 8; ++i) { r[i] = lo[i]; r[i + 8] = hi[i]; }
    return r;
}

__device__ __forceinline__ v8f wmma_f16(v16h a, v16h b, v8f c) {
    return __builtin_amdgcn_wmma_f32_16x16x32_f16(false, a, false, b, (short)0, c, false, false);
}

// ---------------------------------------------------------------------------
// CDNA5 async global->LDS copy (per-lane B128), tracked by ASYNCcnt.
// LDS offset = low 32 bits of generic pointer (aperture maps addr[31:0]).
// ---------------------------------------------------------------------------
__device__ __forceinline__ void async_g2l_b128(unsigned lds_off, const void* gaddr) {
    asm volatile("global_load_async_to_lds_b128 %0, %1, off"
                 :: "v"(lds_off), "v"(gaddr) : "memory");
}
__device__ __forceinline__ void wait_async0() {
    asm volatile("s_wait_asynccnt 0" ::: "memory");
}

// ---------------------------------------------------------------------------
// Tensor Data Mover: 2D tile (tile0 elems/row x tile1 rows, 2-byte elems)
// from strided global tensor into packed LDS. D# per CDNA5 ISA.
// ---------------------------------------------------------------------------
__device__ __forceinline__ void tdm_load_tile_2d_f16(const _Float16* g, unsigned lds_off,
                                                     unsigned tile0, unsigned tile1,
                                                     unsigned long stride0 /*elems*/) {
    unsigned long ga = (unsigned long)g;
    u32x4 g0 = { 1u,                                   // count=1 (valid user D#)
                 lds_off,                               // lds_addr [63:32]
                 (unsigned)ga,                          // global_addr [95:64]
                 (unsigned)((ga >> 32) & 0x01FFFFFFu) | (2u << 30) }; // [120:96] | type=2
    i32x8 g1 = { (int)(1u << 16),                       // data_size=1 (2 bytes)
                 (int)((tile0 & 0xFFFFu) << 16),        // tensor_dim0[15:0] @ [63:48]
                 (int)((tile1 & 0xFFFFu) << 16),        // td0[31:16]=0 | tensor_dim1[15:0]
                 (int)((tile0 & 0xFFFFu) << 16),        // td1[31:16]=0 | tile_dim0 @ [127:112]
                 (int)(tile1 & 0xFFFFu),                // tile_dim1 @ [143:128], tile_dim2=0
                 (int)(unsigned)stride0,                // tensor_dim0_stride[31:0]
                 (int)((stride0 >> 32) & 0xFFFFu),      // stride0[47:32] | td1_stride lo = 0
                 0 };
    i32x4 g2 = { 0, 0, 0, 0 };
    i32x4 g3 = { 0, 0, 0, 0 };
#if defined(__clang_major__) && __clang_major__ >= 23
    i32x8 g4 = { 0, 0, 0, 0, 0, 0, 0, 0 };
    __builtin_amdgcn_tensor_load_to_lds(g0, g1, g2, g3, g4, 0);
#else
    __builtin_amdgcn_tensor_load_to_lds(g0, g1, g2, g3, 0);
#endif
}

// ---------------------------------------------------------------------------
// f32 -> f16 weight conversion
// ---------------------------------------------------------------------------
__global__ void k_cvt_f16(const float* __restrict__ src, _Float16* __restrict__ dst, int n) {
    int i = blockIdx.x * TPB + threadIdx.x;
    if (i < n) dst[i] = (_Float16)src[i];
}

// ---------------------------------------------------------------------------
// h = concat(x, time_features) @ Wp + bp   (K=5 tiny, VALU)
// ---------------------------------------------------------------------------
__global__ void k_embed(const float* __restrict__ x, const float* __restrict__ tf,
                        const float* __restrict__ Wp, const float* __restrict__ bp,
                        float* __restrict__ h) {
    int i = blockIdx.x * TPB + threadIdx.x;   // over 8192*256
    int d = i & 255, t = i >> 8;
    float v = x[t] * Wp[d] + bp[d];
#pragma unroll
    for (int j = 0; j < 4; ++j) v += tf[t * 4 + j] * Wp[(1 + j) * 256 + d];
    h[i] = v;
}

// ---------------------------------------------------------------------------
// LayerNorm over D=256 per token, output f16
// ---------------------------------------------------------------------------
__global__ void k_ln256(const float* __restrict__ h, const float* __restrict__ g,
                        const float* __restrict__ b, _Float16* __restrict__ hn) {
    int t = blockIdx.x, d = threadIdx.x;
    float xv = h[t * 256 + d];
    __shared__ float r[256];
    r[d] = xv; __syncthreads();
#pragma unroll
    for (int s = 128; s > 0; s >>= 1) { if (d < s) r[d] += r[d + s]; __syncthreads(); }
    float mu = r[0] * (1.f / 256.f); __syncthreads();
    r[d] = xv * xv; __syncthreads();
#pragma unroll
    for (int s = 128; s > 0; s >>= 1) { if (d < s) r[d] += r[d + s]; __syncthreads(); }
    float var = r[0] * (1.f / 256.f) - mu * mu;
    hn[t * 256 + d] = (_Float16)((xv - mu) * rsqrtf(var + 1e-5f) * g[d] + b[d]);
}

// ---------------------------------------------------------------------------
// WMMA GEMM: out(M,N) = A(M,K) @ W(K,N) (+bias).  A tile staged via TDM,
// W tile transposed manually into [n][k].  128x128x32 tiles, 8 waves.
// ---------------------------------------------------------------------------
__global__ void k_gemm_f16(const _Float16* __restrict__ A, int lda,
                           const _Float16* __restrict__ W, int N, int Kd,
                           const float* __restrict__ bias,
                           _Float16* __restrict__ outH, float* __restrict__ accF, int ldo) {
    __shared__ __align__(32) _Float16 sA[128 * 32];   // [m][k] (TDM lands packed)
    __shared__ __align__(32) _Float16 sBt[128 * 32];  // [n][k] (transposed)
    int tid = threadIdx.x, w = tid >> 5, ln = tid & 31;
    int bn = blockIdx.x * 128, bm = blockIdx.y * 128;
    int wm = w & 3, wn = w >> 2;           // 4 (M) x 2 (N) wave grid
    int kh = (ln >> 4) * 16;
    unsigned sA_off = (unsigned)(unsigned long)(&sA[0]);

    v8f acc[2][4];
#pragma unroll
    for (int mi = 0; mi < 2; ++mi)
#pragma unroll
        for (int ni = 0; ni < 4; ++ni) acc[mi][ni] = (v8f)(0.f);

    for (int kb = 0; kb < Kd; kb += 32) {
        // A tile via Tensor Data Mover (wave 0 issues one 32x128 tile DMA)
        if (w == 0)
            tdm_load_tile_2d_f16(A + (size_t)bm * lda + kb, sA_off, 32u, 128u,
                                 (unsigned long)lda);
        // B tile: transpose into [n][k]
#pragma unroll
        for (int it = 0; it < 2; ++it) {
            int lin = (tid + it * TPB) * 8;
            int k = lin >> 7, n = lin & 127;
            v8h wv = *(const v8h*)(W + (size_t)(kb + k) * N + bn + n);
#pragma unroll
            for (int j = 0; j < 8; ++j) sBt[(n + j) * 32 + k] = wv[j];
        }
        if (kb + 32 < Kd)
            __builtin_prefetch(W + (size_t)(kb + 32 + (tid & 31)) * N + bn, 0, 0);
        if (w == 0) __builtin_amdgcn_s_wait_tensorcnt(0);
        __syncthreads();

        v16h afrag[2];
#pragma unroll
        for (int mi = 0; mi < 2; ++mi) {
            int row = wm * 32 + mi * 16 + (ln & 15);
            v8h lo = *(const v8h*)&sA[row * 32 + kh];
            v8h hi = *(const v8h*)&sA[row * 32 + kh + 8];
            afrag[mi] = cat8(lo, hi);
        }
#pragma unroll
        for (int ni = 0; ni < 4; ++ni) {
            int col = wn * 64 + ni * 16 + (ln & 15);
            v16h bfrag = *(const v16h*)&sBt[col * 32 + (ln >> 4) * 16];
            acc[0][ni] = wmma_f16(afrag[0], bfrag, acc[0][ni]);
            acc[1][ni] = wmma_f16(afrag[1], bfrag, acc[1][ni]);
        }
        __syncthreads();
    }
#pragma unroll
    for (int mi = 0; mi < 2; ++mi)
#pragma unroll
        for (int ni = 0; ni < 4; ++ni)
#pragma unroll
            for (int j = 0; j < 8; ++j) {
                int row = bm + wm * 32 + mi * 16 + j + ((ln & 16) ? 8 : 0);
                int col = bn + wn * 64 + ni * 16 + (ln & 15);
                float v = acc[mi][ni][j];
                if (bias) v += bias[col];
                if (outH) outH[(size_t)row * ldo + col] = (_Float16)v;
                else      accF[(size_t)row * ldo + col] += v;
            }
}

// ---------------------------------------------------------------------------
// depthwise causal conv (K=4) + SiLU.  xm = up[..., :512] (row stride 1024)
// ---------------------------------------------------------------------------
__global__ void k_conv_silu(const _Float16* __restrict__ up, const float* __restrict__ cw,
                            const float* __restrict__ cb, _Float16* __restrict__ xc) {
    int i = blockIdx.x * TPB + threadIdx.x;     // over 8192*512
    int c = i & 511, t = i >> 9, s = t & 1023, b = t >> 10;
    float a = cb[c];
#pragma unroll
    for (int j = 0; j < 4; ++j) {
        int ss = s - 3 + j;
        if (ss >= 0) a += (float)up[(size_t)(b * 1024 + ss) * 1024 + c] * cw[j * 512 + c];
    }
    xc[i] = (_Float16)(a / (1.f + __expf(-a)));
}

// ---------------------------------------------------------------------------
// gates = [q,k,v] @ Wif + bif  (N=8, VALU dot)
// ---------------------------------------------------------------------------
__global__ void k_gates(const _Float16* __restrict__ q, const _Float16* __restrict__ k,
                        const _Float16* __restrict__ v, const float* __restrict__ Wif,
                        const float* __restrict__ bif, float* __restrict__ gates) {
    int i = blockIdx.x * TPB + threadIdx.x;     // over 8192*8
    int g = i & 7; size_t t = i >> 3;
    const _Float16* qr = q + t * 512; const _Float16* kr = k + t * 512; const _Float16* vr = v + t * 512;
    float acc = bif[g];
    for (int j = 0; j < 512; ++j) acc += (float)qr[j] * Wif[j * 8 + g];
    for (int j = 0; j < 512; ++j) acc += (float)kr[j] * Wif[(512 + j) * 8 + g];
    for (int j = 0; j < 512; ++j) acc += (float)vr[j] * Wif[(1024 + j) * 8 + g];
    gates[t * 8 + g] = acc;
}

// ---------------------------------------------------------------------------
// per (b,h): Fc = cumsum(logsigmoid(f)); Mr[t] = Fc[t] + cummax(i[s]-Fc[s])
// ---------------------------------------------------------------------------
__global__ void k_scan(const float* __restrict__ gates, float* __restrict__ Fc,
                       float* __restrict__ Mr) {
    int bh = blockIdx.x, b = bh >> 2, h = bh & 3, tid = threadIdx.x;
    __shared__ float lf[1024], ib[1024];
    for (int i = tid; i < 1024; i += TPB) {
        float fp = gates[(size_t)(b * 1024 + i) * 8 + 4 + h];
        lf[i] = fminf(fp, 0.f) - log1pf(__expf(-fabsf(fp)));
        ib[i] = gates[(size_t)(b * 1024 + i) * 8 + h];
    }
    __syncthreads();
    if (tid == 0) {
        float fc = 0.f, rm = -1e30f;
        for (int i = 0; i < 1024; ++i) {
            fc += lf[i];
            float a = ib[i] - fc;
            rm = fmaxf(rm, a);
            Fc[bh * 1024 + i] = fc;
            Mr[bh * 1024 + i] = fc + rm;
        }
    }
}

// ---------------------------------------------------------------------------
// mLSTM flash pass: per (b,h, 128-query tile). 64-key tiles, WMMA QK^T & PV.
// K tile staged with CDNA5 async global->LDS copies.
// ---------------------------------------------------------------------------
__global__ void k_mlstm(const _Float16* __restrict__ qb, const _Float16* __restrict__ kb,
                        const _Float16* __restrict__ vb, const float* __restrict__ gates,
                        const float* __restrict__ Fc, const float* __restrict__ Mr,
                        float* __restrict__ ht) {
    __shared__ __align__(32) _Float16 sK[64 * 128];   // [s][d]
    __shared__ __align__(32) _Float16 sVT[128 * 64];  // [d][s]
    __shared__ __align__(32) _Float16 sP[128 * 64];   // [t][s]
    __shared__ float sRowC[128], sEm[128], sSum[128], sColC[64];

    int tid = threadIdx.x, w = tid >> 5, ln = tid & 31;
    int qt = blockIdx.x, bh = blockIdx.y, b = bh >> 2, h = bh & 3;
    int qt0 = qt * 128;
    size_t tok = (size_t)b * 1024;

    if (tid < 128) {
        int t = qt0 + tid;
        float fc = Fc[bh * 1024 + t], m = Mr[bh * 1024 + t];
        sRowC[tid] = fc - m;
        sEm[tid] = __expf(-m);
        sSum[tid] = 0.f;
    }

    // Q fragments (registers): wave w owns rows [w*16, w*16+16)
    const _Float16* qp = qb + (tok + qt0) * 512 + h * 128;
    int qrow = w * 16 + (ln & 15);
    int kh = (ln >> 4) * 16;
    v16h aq[4];
#pragma unroll
    for (int kc = 0; kc < 4; ++kc) {
        v8h lo = *(const v8h*)(qp + (size_t)qrow * 512 + kc * 32 + kh);
        v8h hi = *(const v8h*)(qp + (size_t)qrow * 512 + kc * 32 + kh + 8);
        aq[kc] = cat8(lo, hi);
    }

    v8f hacc[8];
#pragma unroll
    for (int dt = 0; dt < 8; ++dt) hacc[dt] = (v8f)(0.f);

    const float DHI = 0.08838834764831845f;  // 128^-0.5
    int nkt = 2 * qt + 2;
    for (int kt = 0; kt < nkt; ++kt) {
        int s0 = kt * 64;
        // K tile: async global->LDS (per-lane B128, ASYNCcnt)
#pragma unroll
        for (int it = 0; it < 4; ++it) {
            int lin = (tid + it * TPB) * 8;
            int ss = lin >> 7, dd = lin & 127;
            async_g2l_b128((unsigned)(unsigned long)(&sK[lin]),
                           kb + (tok + s0 + ss) * 512 + h * 128 + dd);
        }
        // V tile: transposed [d][s] (manual; async can't transpose)
#pragma unroll
        for (int it = 0; it < 4; ++it) {
            int lin = (tid + it * TPB) * 8;
            int ss = lin >> 7, dd = lin & 127;
            v8h vv = *(const v8h*)(vb + (tok + s0 + ss) * 512 + h * 128 + dd);
#pragma unroll
            for (int j = 0; j < 8; ++j) sVT[(dd + j) * 64 + ss] = vv[j];
        }
        if (tid < 64) {
            int s = s0 + tid;
            sColC[tid] = gates[(tok + s) * 8 + h] - Fc[bh * 1024 + s];
        }
        wait_async0();
        __syncthreads();

        // QK^T : 16 rows x 64 key-cols per wave
        v8f pacc[4];
#pragma unroll
        for (int st = 0; st < 4; ++st) pacc[st] = (v8f)(0.f);
#pragma unroll
        for (int st = 0; st < 4; ++st) {
            int scol = st * 16 + (ln & 15);
#pragma unroll
            for (int kc = 0; kc < 4; ++kc) {
                v16h bk = *(const v16h*)&sK[scol * 128 + kc * 32 + (ln >> 4) * 16];
                pacc[st] = wmma_f16(aq[kc], bk, pacc[st]);
            }
        }

        // decay weights + causal mask + row-sum + stage P (f16) in LDS
#pragma unroll
        for (int st = 0; st < 4; ++st)
#pragma unroll
            for (int j = 0; j < 8; ++j) {
                int tl = w * 16 + j + ((ln & 16) ? 8 : 0);
                int sl = st * 16 + (ln & 15);
                float p = 0.f;
                if (s0 + sl <= qt0 + tl)
                    p = pacc[st][j] * DHI * __expf(sRowC[tl] + sColC[sl]);
                atomicAdd(&sSum[tl], p);
                sP[tl * 64 + sl] = (_Float16)p;
            }
        __syncthreads();

        // PV : hacc(16x128) += P(16x64) @ V(64x128)
        int prow = w * 16 + (ln & 15);
#pragma unroll
        for (int sc = 0; sc < 2; ++sc) {
            v8h lo = *(const v8h*)&sP[prow * 64 + sc * 32 + kh];
            v8h hi = *(const v8h*)&sP[prow * 64 + sc * 32 + kh + 8];
            v16h ap = cat8(lo, hi);
#pragma unroll
            for (int dt = 0; dt < 8; ++dt) {
                int dcol = dt * 16 + (ln & 15);
                v16h bv = *(const v16h*)&sVT[dcol * 64 + sc * 32 + (ln >> 4) * 16];
                hacc[dt] = wmma_f16(ap, bv, hacc[dt]);
            }
        }
        __syncthreads();
    }

    // normalize + write
#pragma unroll
    for (int dt = 0; dt < 8; ++dt)
#pragma unroll
        for (int j = 0; j < 8; ++j) {
            int tl = w * 16 + j + ((ln & 16) ? 8 : 0);
            int dcol = dt * 16 + (ln & 15);
            float den = fmaxf(fabsf(sSum[tl]), sEm[tl]) + 1e-6f;
            ht[(tok + qt0 + tl) * 512 + h * 128 + dcol] = hacc[dt][j] / den;
        }
}

// ---------------------------------------------------------------------------
// per-head group norm + skip*xc + SiLU(z) gate -> f16
// ---------------------------------------------------------------------------
__global__ void k_gn_gate(const float* __restrict__ ht, const float* __restrict__ gg,
                          const float* __restrict__ gb, const float* __restrict__ skip,
                          const _Float16* __restrict__ xc, const _Float16* __restrict__ up,
                          _Float16* __restrict__ htg) {
    int blk = blockIdx.x;                // over 8192*4
    size_t t = blk >> 2; int h = blk & 3, d = threadIdx.x;
    int c = h * 128 + d;
    float xv = ht[t * 512 + c];
    __shared__ float r[128];
    r[d] = xv; __syncthreads();
#pragma unroll
    for (int s = 64; s > 0; s >>= 1) { if (d < s) r[d] += r[d + s]; __syncthreads(); }
    float mu = r[0] * (1.f / 128.f); __syncthreads();
    r[d] = xv * xv; __syncthreads();
#pragma unroll
    for (int s = 64; s > 0; s >>= 1) { if (d < s) r[d] += r[d + s]; __syncthreads(); }
    float var = r[0] * (1.f / 128.f) - mu * mu;
    float y = (xv - mu) * rsqrtf(var + 1e-5f) * gg[c] + gb[c];
    y += skip[c] * (float)xc[t * 512 + c];
    float z = (float)up[t * 1024 + 512 + c];
    y *= z / (1.f + __expf(-z));
    htg[t * 512 + c] = (_Float16)y;
}

// ---------------------------------------------------------------------------
// final LN + head:  out[b] = LN(h[b, S-1, :]) @ Wf + bf
// ---------------------------------------------------------------------------
__global__ void k_final(const float* __restrict__ h, const float* __restrict__ g,
                        const float* __restrict__ bt, const float* __restrict__ Wf,
                        const float* __restrict__ bf, float* __restrict__ out) {
    int b = blockIdx.x, d = threadIdx.x;
    size_t t = (size_t)b * 1024 + 1023;
    float xv = h[t * 256 + d];
    __shared__ float r[256];
    r[d] = xv; __syncthreads();
#pragma unroll
    for (int s = 128; s > 0; s >>= 1) { if (d < s) r[d] += r[d + s]; __syncthreads(); }
    float mu = r[0] * (1.f / 256.f); __syncthreads();
    r[d] = xv * xv; __syncthreads();
#pragma unroll
    for (int s = 128; s > 0; s >>= 1) { if (d < s) r[d] += r[d + s]; __syncthreads(); }
    float var = r[0] * (1.f / 256.f) - mu * mu; __syncthreads();
    float hn = (xv - mu) * rsqrtf(var + 1e-5f) * g[d] + bt[d];
    r[d] = hn * Wf[d]; __syncthreads();
#pragma unroll
    for (int s = 128; s > 0; s >>= 1) { if (d < s) r[d] += r[d + s]; __syncthreads(); }
    if (d == 0) out[b] = r[0] + bf[0];
}

// ---------------------------------------------------------------------------
extern "C" void kernel_launch(void* const* d_in, const int* in_sizes, int n_in,
                              void* d_out, int out_size, void* d_ws, size_t ws_size,
                              hipStream_t stream) {
    const float* x    = (const float*)d_in[0];
    const float* tf   = (const float*)d_in[1];
    const float* Wp   = (const float*)d_in[2];
    const float* bp   = (const float*)d_in[3];
    const float* ln_g = (const float*)d_in[4];
    const float* ln_b = (const float*)d_in[5];
    const float* Wup  = (const float*)d_in[6];
    const float* bup  = (const float*)d_in[7];
    const float* cw   = (const float*)d_in[8];
    const float* cb   = (const float*)d_in[9];
    const float* Wq   = (const float*)d_in[10];
    const float* Wk   = (const float*)d_in[11];
    const float* Wv   = (const float*)d_in[12];
    const float* Wif  = (const float*)d_in[13];
    const float* bif  = (const float*)d_in[14];
    const float* gn_g = (const float*)d_in[15];
    const float* gn_b = (const float*)d_in[16];
    const float* skip = (const float*)d_in[17];
    const float* Wdn  = (const float*)d_in[18];
    const float* bdn  = (const float*)d_in[19];
    const float* lnfg = (const float*)d_in[20];
    const float* lnfb = (const float*)d_in[21];
    const float* Wf   = (const float*)d_in[22];
    const float* bf   = (const float*)d_in[23];
    float* out = (float*)d_out;

    const int M = 8192;  // B*S
    char* base = (char*)d_ws; size_t off = 0;
    auto alloc = [&](size_t bytes) -> void* {
        off = (off + 255) & ~(size_t)255;
        void* p = base + off; off += bytes; return p;
    };
    float*    hbuf = (float*)   alloc((size_t)M * 256 * 4);
    _Float16* hn   = (_Float16*)alloc((size_t)M * 256 * 2);
    _Float16* up   = (_Float16*)alloc((size_t)M * 1024 * 2);
    _Float16* xc   = (_Float16*)alloc((size_t)M * 512 * 2);
    _Float16* qbf  = (_Float16*)alloc((size_t)M * 512 * 2);
    _Float16* kbf  = (_Float16*)alloc((size_t)M * 512 * 2);
    _Float16* vbf  = (_Float16*)alloc((size_t)M * 512 * 2);
    float*    gts  = (float*)   alloc((size_t)M * 8 * 4);
    float*    FcB  = (float*)   alloc((size_t)32 * 1024 * 4);
    float*    MrB  = (float*)   alloc((size_t)32 * 1024 * 4);
    float*    htb  = (float*)   alloc((size_t)M * 512 * 4);
    _Float16* htg  = (_Float16*)alloc((size_t)M * 512 * 2);
    _Float16* wUp16 = (_Float16*)alloc((size_t)4 * 256 * 1024 * 2);
    _Float16* wQ16  = (_Float16*)alloc((size_t)4 * 512 * 512 * 2);
    _Float16* wK16  = (_Float16*)alloc((size_t)4 * 512 * 512 * 2);
    _Float16* wV16  = (_Float16*)alloc((size_t)4 * 512 * 512 * 2);
    _Float16* wD16  = (_Float16*)alloc((size_t)4 * 512 * 256 * 2);

    k_cvt_f16<<<(4 * 256 * 1024) / TPB, TPB, 0, stream>>>(Wup, wUp16, 4 * 256 * 1024);
    k_cvt_f16<<<(4 * 512 * 512) / TPB, TPB, 0, stream>>>(Wq, wQ16, 4 * 512 * 512);
    k_cvt_f16<<<(4 * 512 * 512) / TPB, TPB, 0, stream>>>(Wk, wK16, 4 * 512 * 512);
    k_cvt_f16<<<(4 * 512 * 512) / TPB, TPB, 0, stream>>>(Wv, wV16, 4 * 512 * 512);
    k_cvt_f16<<<(4 * 512 * 256) / TPB, TPB, 0, stream>>>(Wdn, wD16, 4 * 512 * 256);

    k_embed<<<(M * 256) / TPB, TPB, 0, stream>>>(x, tf, Wp, bp, hbuf);

    for (int l = 0; l < 4; ++l) {
        k_ln256<<<M, 256, 0, stream>>>(hbuf, ln_g + l * 256, ln_b + l * 256, hn);
        k_gemm_f16<<<dim3(1024 / 128, M / 128), TPB, 0, stream>>>(
            hn, 256, wUp16 + (size_t)l * 256 * 1024, 1024, 256,
            bup + l * 1024, up, nullptr, 1024);
        k_conv_silu<<<(M * 512) / TPB, TPB, 0, stream>>>(up, cw + l * 4 * 512, cb + l * 512, xc);
        k_gemm_f16<<<dim3(512 / 128, M / 128), TPB, 0, stream>>>(
            xc, 512, wQ16 + (size_t)l * 512 * 512, 512, 512, nullptr, qbf, nullptr, 512);
        k_gemm_f16<<<dim3(512 / 128, M / 128), TPB, 0, stream>>>(
            xc, 512, wK16 + (size_t)l * 512 * 512, 512, 512, nullptr, kbf, nullptr, 512);
        k_gemm_f16<<<dim3(512 / 128, M / 128), TPB, 0, stream>>>(
            up, 1024, wV16 + (size_t)l * 512 * 512, 512, 512, nullptr, vbf, nullptr, 512);
        k_gates<<<(M * 8) / TPB, TPB, 0, stream>>>(qbf, kbf, vbf, Wif + (size_t)l * 1536 * 8,
                                                   bif + l * 8, gts);
        k_scan<<<32, TPB, 0, stream>>>(gts, FcB, MrB);
        k_mlstm<<<dim3(8, 32), TPB, 0, stream>>>(qbf, kbf, vbf, gts, FcB, MrB, htb);
        k_gn_gate<<<M * 4, 128, 0, stream>>>(htb, gn_g + l * 512, gn_b + l * 512,
                                             skip + l * 512, xc, up, htg);
        k_gemm_f16<<<dim3(256 / 128, M / 128), TPB, 0, stream>>>(
            htg, 512, wD16 + (size_t)l * 512 * 256, 256, 512,
            bdn + l * 256, nullptr, hbuf, 256);
    }
    k_final<<<8, 256, 0, stream>>>(hbuf, lnfg, lnfb, Wf, bf, out);
}
